// WaveNet_31731218383198
// MI455X (gfx1250) — matrix operand order repaired and checked by reading, not compiled, not run on hardware
//
#include <hip/hip_runtime.h>
#include <hip/hip_bf16.h>

// ---------------------------------------------------------------------------
// WaveNet autoregressive sampler for MI455X (gfx1250).
// Latency-bound sequential problem -> ONE persistent workgroup (32 wave32s on
// one WGP), full t-loop + 30-layer chain inside the kernel, activations in
// LDS, bf16 weights fragment-packed for v_wmma_f32_16x16x32_bf16 (f32 acc).
// Weights (~7MB bf16) are L2-resident; next layer's fragments are warmed into
// the WGP$ with global_prefetch while the current layer computes; the WMMA
// K-loop is software-pipelined (double-buffered fragments) so global/ds loads
// overlap the matrix ops.
// ---------------------------------------------------------------------------

#define CCH   128   // C
#define SCH   256   // S
#define NCAT  256   // CAT
#define BATCH 32
#define LAY   30
#define NTHREADS 1024        // == BATCH*CCH/4, one float4 of x per thread
#define NWAVES (NTHREADS / 32)

typedef __attribute__((ext_vector_type(16))) __bf16        v16bf;
typedef __attribute__((ext_vector_type(8)))  float         v8f;
typedef __attribute__((ext_vector_type(4)))  unsigned int  v4u;

union Frag16 { v16bf v; v4u q[2]; };

__device__ __forceinline__ unsigned short f2bf(float f) {
  unsigned int u = __float_as_uint(f);
  u += 0x7fffu + ((u >> 16) & 1u);          // round-to-nearest-even
  return (unsigned short)(u >> 16);
}
__device__ __forceinline__ unsigned int pack2bf(float a, float b) {
  return (unsigned int)f2bf(a) | ((unsigned int)f2bf(b) << 16);
}

// ---------------------------------------------------------------------------
// Pack row-major [Kdim][N] f32 matrices (blockIdx.y = matrix index) into the
// bf16 WMMA B-matrix fragment layout used here, contiguous k order (same as A):
//   lane 0-15 : N = nt*16 + lane,      elements j=0..15 hold K = kt*32 + j
//   lane16-31 : N = nt*16 + (lane-16), elements j=0..15 hold K = kt*32+16+j
// Fragment f = (kt*ntiles + nt); storage dst[f*512 + lane*16 + j].
// ---------------------------------------------------------------------------
__global__ void pack_frags(const float* __restrict__ src,
                           unsigned short* __restrict__ dst,
                           int Kdim, int N) {
  int mat = blockIdx.y;
  long long base = (long long)mat * Kdim * N;
  int ntiles = N >> 4;
  for (int idx = blockIdx.x * blockDim.x + threadIdx.x; idx < Kdim * N;
       idx += gridDim.x * blockDim.x) {
    int k = idx / N, n = idx - k * N;
    int kt = k >> 5, kr = k & 31;
    int nt = n >> 4, nr = n & 15;
    int lane = nr + ((kr >> 4) << 4);
    int j = kr & 15;
    dst[base + ((long long)(kt * ntiles + nt) << 9) + lane * 16 + j] =
        f2bf(src[base + idx]);
  }
}

__global__ void zero_ws(float* p, long long n) {
  for (long long i = blockIdx.x * (long long)blockDim.x + threadIdx.x; i < n;
       i += (long long)gridDim.x * blockDim.x)
    p[i] = 0.f;
}

// ---------------------------------------------------------------------------
// One 16x16 C-tile, K = ktiles*32. A = bf16 row-major in LDS (stride lda),
// B = fragment-packed bf16 in global. Software-pipelined: fragments for kt+1
// are issued before the WMMA for kt, so the v_wmma only needs partial waits
// (in-order load/ds return) and load latency overlaps the matrix pipe.
// ---------------------------------------------------------------------------
__device__ __forceinline__ v8f tile_mma(const unsigned short* __restrict__ Abase,
                                        int lda,
                                        const unsigned short* __restrict__ Bf,
                                        int mt, int nt, int ktiles, int ntiles,
                                        int lane) {
  v8f acc = {0.f, 0.f, 0.f, 0.f, 0.f, 0.f, 0.f, 0.f};
  const unsigned short* row =
      Abase + (mt * 16 + (lane & 15)) * lda + ((lane >> 4) << 4);
  const unsigned short* bbase = Bf + (((long long)nt << 5) + lane) * 16;
  const long long bstride = (long long)ntiles << 9;   // elems per kt step

  Frag16 a[2], b[2];
  a[0].q[0] = *(const v4u*)(row);
  a[0].q[1] = *(const v4u*)(row + 8);
  {
    const v4u* bp = (const v4u*)bbase;
    b[0].q[0] = bp[0];
    b[0].q[1] = bp[1];
  }
#pragma unroll
  for (int kt = 0; kt < ktiles; ++kt) {
    const int cur = kt & 1, nxt = cur ^ 1;
    if (kt + 1 < ktiles) {
      const unsigned short* ap = row + (kt + 1) * 32;
      a[nxt].q[0] = *(const v4u*)(ap);
      a[nxt].q[1] = *(const v4u*)(ap + 8);
      const v4u* bp = (const v4u*)(bbase + (long long)(kt + 1) * bstride);
      b[nxt].q[0] = bp[0];
      b[nxt].q[1] = bp[1];
    }
    acc = __builtin_amdgcn_wmma_f32_16x16x32_bf16(false, a[cur].v, false,
                                                  b[cur].v, (short)0, acc,
                                                  false, false);
  }
  return acc;
}

// C/D layout (ISA 7.12.2): VGPR r -> M = mt*16 + r + (lane>=16 ? 8 : 0),
// N = nt*16 + (lane&15).

__global__ __launch_bounds__(NTHREADS)
void wavenet_run(const int* __restrict__ seed,
                 const float* __restrict__ k_init, const float* __restrict__ b_init,
                 const float* __restrict__ b_conv, const float* __restrict__ br,
                 const float* __restrict__ bs, const float* __restrict__ b0,
                 const float* __restrict__ b1,
                 const unsigned short* __restrict__ fK,
                 const unsigned short* __restrict__ fWr,
                 const unsigned short* __restrict__ fWs,
                 const unsigned short* __restrict__ fW0,
                 const unsigned short* __restrict__ fW1,
                 float* __restrict__ queues,
                 float* __restrict__ out_pred, float* __restrict__ out_samp,
                 float* __restrict__ out_logits, int T) {
  extern __shared__ char smem[];
  float* s_x    = (float*)smem;                                  // 32x128 f32 (residual x)
  float* s_f    = s_x + BATCH * CCH;                             // 32x256 f32 (h / logits)
  float* s_skip = s_f + BATCH * 256;                             // 32x256 f32
  unsigned short* s_a = (unsigned short*)(s_skip + BATCH * 256); // 32x256 bf16 A-operand
  unsigned short* s_b = s_a + BATCH * 256;                       // 32x256 bf16 (h0)
  unsigned short* s_g = s_b + BATCH * 256;                       // 32x128 bf16 (gate out)
  __shared__ int s_cur[BATCH], s_last[BATCH];

  const int tid  = threadIdx.x;
  const int lane = tid & 31;
  const int wid  = tid >> 5;

  if (tid < BATCH) { s_cur[tid] = seed[tid]; s_last[tid] = seed[tid]; }
  __syncthreads();

  for (int t = 0; t < T; ++t) {
    // embedding: x = k_init[0][last] + k_init[1][cur] + b_init ; skip = 0
    {
      const int i = tid;                 // 1024 float4 slots == 32x128 floats
      const int b = i >> 5, c = (i & 31) << 2;
      const float4 e0 = *(const float4*)(k_init + s_last[b] * CCH + c);
      const float4 e1 = *(const float4*)(k_init + NCAT * CCH + s_cur[b] * CCH + c);
      const float4 bi = *(const float4*)(b_init + c);
      float4 r;
      r.x = e0.x + e1.x + bi.x; r.y = e0.y + e1.y + bi.y;
      r.z = e0.z + e1.z + bi.z; r.w = e0.w + e1.w + bi.w;
      ((float4*)s_x)[i] = r;
      const float4 z = {0.f, 0.f, 0.f, 0.f};
      ((float4*)s_skip)[i]            = z;   // 2048 float4s total
      ((float4*)s_skip)[i + NTHREADS] = z;
    }
    __syncthreads();

    long long qbase = 0;
    for (int li = 0; li < LAY; ++li) {
      int d = 1 << (li % 10);
      long long qoff = qbase + (long long)(t & (d - 1)) * (BATCH * CCH);
      // dilation queue swap (b128 each way); build A = [x_last | x] as bf16
      {
        const int i = tid;               // one float4 per thread
        float4 xl = ((const float4*)(queues + qoff))[i];
        float4 xc = ((const float4*)s_x)[i];
        ((float4*)(queues + qoff))[i] = xc;
        const int b = i >> 5, c = (i & 31) << 2;
        unsigned int* al = (unsigned int*)(s_a + b * 256 + c);
        al[0] = pack2bf(xl.x, xl.y);
        al[1] = pack2bf(xl.z, xl.w);
        unsigned int* ac = (unsigned int*)(s_a + b * 256 + 128 + c);
        ac[0] = pack2bf(xc.x, xc.y);
        ac[1] = pack2bf(xc.z, xc.w);
      }
      // warm next layer's weight fragments into WGP$ (global_prefetch),
      // 128B per slot: conv 1024 slots, Wr 256, Ws 512.
      if (li + 1 < LAY) {
        const char* pK = (const char*)(fK  + (long long)(li + 1) * 256 * 256);
        const char* pR = (const char*)(fWr + (long long)(li + 1) * CCH * CCH);
        const char* pS = (const char*)(fWs + (long long)(li + 1) * CCH * SCH);
        for (int s = tid; s < 1792; s += NTHREADS) {
          const char* p;
          if (s < 1024)      p = pK + ((size_t)s << 7);
          else if (s < 1280) p = pR + ((size_t)(s - 1024) << 7);
          else               p = pS + ((size_t)(s - 1280) << 7);
          __builtin_prefetch(p, 0, 1);
        }
      }
      __syncthreads();

      // conv: h = [x_last|x] @ K[li] + b_conv   ([32x256] @ [256x256])
      const unsigned short* Bf = fK + (long long)li * 256 * 256;
      for (int tile = wid; tile < 32; tile += NWAVES) {
        int mt = tile >> 4, nt = tile & 15;
        v8f acc = tile_mma(s_a, 256, Bf, mt, nt, 8, 16, lane);
        int n  = nt * 16 + (lane & 15);
        int mb = mt * 16 + ((lane >> 4) << 3);
        float bias = b_conv[li * 256 + n];
        for (int r = 0; r < 8; ++r) s_f[(mb + r) * 256 + n] = acc[r] + bias;
      }
      __syncthreads();

      // gate: g = tanh(h[:, :C]) * sigmoid(h[:, C:])
      for (int i = tid; i < BATCH * CCH; i += NTHREADS) {
        int b = i >> 7, c = i & 127;
        float th = tanhf(s_f[b * 256 + c]);
        float sg = 1.f / (1.f + __expf(-s_f[b * 256 + 128 + c]));
        s_g[i] = f2bf(th * sg);
      }
      __syncthreads();

      // residual x += g@Wr + br  (16 tiles) ; skip += g@Ws + bs (32 tiles)
      const unsigned short* BfR = fWr + (long long)li * CCH * CCH;
      const unsigned short* BfS = fWs + (long long)li * CCH * SCH;
      for (int tile = wid; tile < 48; tile += NWAVES) {
        if (tile < 16) {
          int mt = tile >> 3, nt = tile & 7;
          v8f acc = tile_mma(s_g, 128, BfR, mt, nt, 4, 8, lane);
          int n  = nt * 16 + (lane & 15);
          int mb = mt * 16 + ((lane >> 4) << 3);
          float bias = br[li * CCH + n];
          for (int r = 0; r < 8; ++r) s_x[(mb + r) * 128 + n] += acc[r] + bias;
        } else {
          int t2 = tile - 16;
          int mt = t2 >> 4, nt = t2 & 15;
          v8f acc = tile_mma(s_g, 128, BfS, mt, nt, 4, 16, lane);
          int n  = nt * 16 + (lane & 15);
          int mb = mt * 16 + ((lane >> 4) << 3);
          float bias = bs[li * SCH + n];
          for (int r = 0; r < 8; ++r) s_skip[(mb + r) * 256 + n] += acc[r] + bias;
        }
      }
      __syncthreads();
      qbase += (long long)d * (BATCH * CCH);
    }

    // head: relu(skip) -> bf16
    for (int i = tid; i < BATCH * SCH; i += NTHREADS) {
      float v = s_skip[i];
      s_a[i] = f2bf(v > 0.f ? v : 0.f);
    }
    __syncthreads();
    // h0 = relu(relu(skip) @ W0 + b0) -> bf16
    for (int tile = wid; tile < 32; tile += NWAVES) {
      int mt = tile >> 4, nt = tile & 15;
      v8f acc = tile_mma(s_a, 256, fW0, mt, nt, 8, 16, lane);
      int n  = nt * 16 + (lane & 15);
      int mb = mt * 16 + ((lane >> 4) << 3);
      float bias = b0[n];
      for (int r = 0; r < 8; ++r) {
        float v = acc[r] + bias;
        s_b[(mb + r) * 256 + n] = f2bf(v > 0.f ? v : 0.f);
      }
    }
    __syncthreads();
    // logits = h0 @ W1 + b1
    for (int tile = wid; tile < 32; tile += NWAVES) {
      int mt = tile >> 4, nt = tile & 15;
      v8f acc = tile_mma(s_b, 256, fW1, mt, nt, 8, 16, lane);
      int n  = nt * 16 + (lane & 15);
      int mb = mt * 16 + ((lane >> 4) << 3);
      float bias = b1[n];
      for (int r = 0; r < 8; ++r) s_f[(mb + r) * 256 + n] = acc[r] + bias;
    }
    __syncthreads();

    // write logits; argmax per batch row; mu-law expand; feed sample back
    for (int i = tid; i < BATCH * NCAT; i += NTHREADS) {
      int b = i >> 8, n = i & 255;
      out_logits[((long long)b * T + t) * NCAT + n] = s_f[i];
    }
    if (tid < BATCH) {
      const float* row = s_f + tid * NCAT;
      float best = row[0];
      int arg = 0;
      for (int n = 1; n < NCAT; ++n) {
        float v = row[n];
        if (v > best) { best = v; arg = n; }
      }
      out_samp[tid * T + t] = (float)arg;
      float miu = 2.f * (float)arg / 255.f - 1.f;
      float mag = (powf(256.f, fabsf(miu)) - 1.f) / 255.f;
      out_pred[tid * T + t] = (miu > 0.f) ? mag : ((miu < 0.f) ? -mag : 0.f);
      s_last[tid] = s_cur[tid];
      s_cur[tid]  = arg;
    }
    __syncthreads();
  }
}

// ---------------------------------------------------------------------------
extern "C" void kernel_launch(void* const* d_in, const int* in_sizes, int n_in,
                              void* d_out, int out_size, void* d_ws, size_t ws_size,
                              hipStream_t stream) {
  const int*   seed   = (const int*)d_in[0];
  const float* k_init = (const float*)d_in[2];
  const float* b_init = (const float*)d_in[3];
  const float* K      = (const float*)d_in[4];   // [L][2][C][2C] == [L][256][256]
  const float* b_conv = (const float*)d_in[5];
  const float* Wr     = (const float*)d_in[6];
  const float* br     = (const float*)d_in[7];
  const float* Ws     = (const float*)d_in[8];
  const float* bs     = (const float*)d_in[9];
  const float* W0     = (const float*)d_in[10];
  const float* b0     = (const float*)d_in[11];
  const float* W1     = (const float*)d_in[12];
  const float* b1     = (const float*)d_in[13];
  (void)in_sizes; (void)n_in; (void)ws_size;

  int T = out_size / (BATCH * (2 + NCAT));       // pred + samples + logits

  // Workspace: dilation queues (sum(d)=3069 slots of [32][128] f32) + bf16 frags
  const long long QF = 3069LL * BATCH * CCH;
  char* ws = (char*)d_ws;
  size_t off = 0;
  float* queues = (float*)(ws + off); off += (size_t)QF * 4;
  off = (off + 255) & ~(size_t)255;
  unsigned short* fK  = (unsigned short*)(ws + off); off += (size_t)LAY * 256 * 256 * 2;
  off = (off + 255) & ~(size_t)255;
  unsigned short* fWr = (unsigned short*)(ws + off); off += (size_t)LAY * CCH * CCH * 2;
  off = (off + 255) & ~(size_t)255;
  unsigned short* fWs = (unsigned short*)(ws + off); off += (size_t)LAY * CCH * SCH * 2;
  off = (off + 255) & ~(size_t)255;
  unsigned short* fW0 = (unsigned short*)(ws + off); off += (size_t)SCH * NCAT * 2;
  off = (off + 255) & ~(size_t)255;
  unsigned short* fW1 = (unsigned short*)(ws + off); off += (size_t)NCAT * NCAT * 2;

  zero_ws<<<2048, 256, 0, stream>>>(queues, QF);
  pack_frags<<<dim3(256, LAY), 256, 0, stream>>>(K,  fK,  256, 256);
  pack_frags<<<dim3(64,  LAY), 256, 0, stream>>>(Wr, fWr, 128, 128);
  pack_frags<<<dim3(128, LAY), 256, 0, stream>>>(Ws, fWs, 128, 256);
  pack_frags<<<dim3(256, 1),   256, 0, stream>>>(W0, fW0, 256, 256);
  pack_frags<<<dim3(256, 1),   256, 0, stream>>>(W1, fW1, 256, 256);

  float* out = (float*)d_out;
  size_t smem = (size_t)(BATCH * CCH + 2 * BATCH * 256) * 4 +
                (size_t)(2 * BATCH * 256 + BATCH * CCH) * 2;   // ~120KB (<320KB/WGP)
  wavenet_run<<<1, NTHREADS, smem, stream>>>(
      seed, k_init, b_init, b_conv, br, bs, b0, b1,
      fK, fWr, fWs, fW0, fW1, queues,
      out, out + (size_t)BATCH * T, out + (size_t)2 * BATCH * T, T);
}
